// MaskedMultiheadSelfAttention_13348758356039
// MI455X (gfx1250) — matrix operand order repaired and checked by reading, not compiled, not run on hardware
//
#include <hip/hip_runtime.h>
#include <math.h>

// ---------------------------------------------------------------------------
// MI455X (gfx1250) masked multi-head self-attention, bf16 WMMA + fp32 accum.
// B=2, T=2048, C=1024, H=16, D=64
// Ping-pong software-pipelined GEMMs, flash attention with K prefetch.
// ---------------------------------------------------------------------------

#define BB 2
#define TT 2048
#define CC 1024
#define HH 16
#define DD 64
#define MTOT (BB * TT)          // 4096
#define N_QKV (3 * CC)          // 3072

typedef __attribute__((ext_vector_type(16))) __bf16 v16bf;
typedef __attribute__((ext_vector_type(8)))  __bf16 v8bf;
typedef __attribute__((ext_vector_type(8)))  float  v8f;

union V16U { v16bf v; v8bf h[2]; };

#define WMMA_BF16(A, B, C) \
  __builtin_amdgcn_wmma_f32_16x16x32_bf16(false, (A), false, (B), (short)0, (C), false, false)

__device__ __forceinline__ unsigned short f2bf(float f) {
  union { float f; unsigned int u; } v; v.f = f;
  unsigned int u = v.u;
  u += 0x7fffu + ((u >> 16) & 1u);     // round-to-nearest-even
  return (unsigned short)(u >> 16);
}

// Load a 16-element bf16 fragment (two 16-byte halves) from two pointers.
__device__ __forceinline__ v16bf load_frag16(const unsigned short* p0,
                                             const unsigned short* p1) {
  V16U u;
  u.h[0] = *reinterpret_cast<const v8bf*>(p0);
  u.h[1] = *reinterpret_cast<const v8bf*>(p1);
  return u.v;
}

// ---------------------------------------------------------------------------
// fp32 -> bf16 conversion (4 elements / thread)
// ---------------------------------------------------------------------------
__global__ __launch_bounds__(256)
void cvt_f32_bf16(const float* __restrict__ src, unsigned short* __restrict__ dst, int n) {
  int i = (blockIdx.x * blockDim.x + threadIdx.x) * 4;
  if (i < n) {
    float4 f = *reinterpret_cast<const float4*>(src + i);
    ushort4 o;
    o.x = f2bf(f.x); o.y = f2bf(f.y); o.z = f2bf(f.z); o.w = f2bf(f.w);
    *reinterpret_cast<ushort4*>(dst + i) = o;
  }
}

// ---------------------------------------------------------------------------
// Core pipelined 32M x 64N wave-tile GEMM over K = CC (both operands
// K-contiguous bf16). Ping-pong fragment sets: no rotation copies, no
// WMMA->VALU WAR hazards on fragment registers.
// ---------------------------------------------------------------------------
__device__ __forceinline__ void mma_tile_32x64(
    const unsigned short* __restrict__ a0row,   // row mBase    + l15
    const unsigned short* __restrict__ a1row,   // row mBase+16 + l15
    const unsigned short* __restrict__ b0row,   // row nBase       + l15
    const unsigned short* __restrict__ b1row,   // row nBase+16    + l15
    const unsigned short* __restrict__ b2row,
    const unsigned short* __restrict__ b3row,
    int kh, v8f acc[2][4]) {
  auto loadA = [&](const unsigned short* p, int kb) {
    return load_frag16(p + kb + kh * 16, p + kb + kh * 16 + 8);
  };
  auto loadB = [&](const unsigned short* p, int kb) {
    return load_frag16(p + kb + kh * 16, p + kb + kh * 16 + 8);
  };

  v16bf aP[2], bP[4], aQ[2], bQ[4];
  aP[0] = loadA(a0row, 0); aP[1] = loadA(a1row, 0);
  bP[0] = loadB(b0row, 0); bP[1] = loadB(b1row, 0);
  bP[2] = loadB(b2row, 0); bP[3] = loadB(b3row, 0);

  int kb = 0;
  for (; kb < CC - 64; kb += 64) {
    // prefetch set Q at kb+32, compute set P (k = kb)
    aQ[0] = loadA(a0row, kb + 32); aQ[1] = loadA(a1row, kb + 32);
    bQ[0] = loadB(b0row, kb + 32); bQ[1] = loadB(b1row, kb + 32);
    bQ[2] = loadB(b2row, kb + 32); bQ[3] = loadB(b3row, kb + 32);
#pragma unroll
    for (int mi = 0; mi < 2; ++mi)
#pragma unroll
      for (int t = 0; t < 4; ++t)
        acc[mi][t] = WMMA_BF16(aP[mi], bP[t], acc[mi][t]);

    // prefetch set P at kb+64, compute set Q (k = kb+32)
    aP[0] = loadA(a0row, kb + 64); aP[1] = loadA(a1row, kb + 64);
    bP[0] = loadB(b0row, kb + 64); bP[1] = loadB(b1row, kb + 64);
    bP[2] = loadB(b2row, kb + 64); bP[3] = loadB(b3row, kb + 64);
#pragma unroll
    for (int mi = 0; mi < 2; ++mi)
#pragma unroll
      for (int t = 0; t < 4; ++t)
        acc[mi][t] = WMMA_BF16(aQ[mi], bQ[t], acc[mi][t]);
  }

  // kb == CC-64: prefetch Q at CC-32, compute P, then compute Q
  aQ[0] = loadA(a0row, CC - 32); aQ[1] = loadA(a1row, CC - 32);
  bQ[0] = loadB(b0row, CC - 32); bQ[1] = loadB(b1row, CC - 32);
  bQ[2] = loadB(b2row, CC - 32); bQ[3] = loadB(b3row, CC - 32);
#pragma unroll
  for (int mi = 0; mi < 2; ++mi)
#pragma unroll
    for (int t = 0; t < 4; ++t)
      acc[mi][t] = WMMA_BF16(aP[mi], bP[t], acc[mi][t]);
#pragma unroll
  for (int mi = 0; mi < 2; ++mi)
#pragma unroll
    for (int t = 0; t < 4; ++t)
      acc[mi][t] = WMMA_BF16(aQ[mi], bQ[t], acc[mi][t]);
}

// ---------------------------------------------------------------------------
// QKV projection: y[m, o] = sum_c x[m,c] * Wqkv[o,c] + b[o]
// M=4096, N=3072, K=1024. Wave tile: 32M x 64N. Block: 8 waves = 64M x 256N.
// Scatter into Q,K as (B,H,T,D) bf16, V transposed as (B,H,D,T) bf16.
// ---------------------------------------------------------------------------
__global__ __launch_bounds__(256)
void qkv_gemm(const unsigned short* __restrict__ xb,
              const unsigned short* __restrict__ wb,
              const float* __restrict__ bias,
              unsigned short* __restrict__ Qw,
              unsigned short* __restrict__ Kw,
              unsigned short* __restrict__ Vt) {
  const int tid  = threadIdx.x;
  const int w    = tid >> 5;
  const int lane = tid & 31;
  const int l15  = lane & 15;
  const int kh   = lane >> 4;
  const int nBase = blockIdx.x * 256 + (w >> 1) * 64;
  const int mBase = blockIdx.y * 64  + (w & 1) * 32;

  v8f acc[2][4] = {};
  mma_tile_32x64(xb + (size_t)(mBase + l15) * CC,
                 xb + (size_t)(mBase + 16 + l15) * CC,
                 wb + (size_t)(nBase +  0 + l15) * CC,
                 wb + (size_t)(nBase + 16 + l15) * CC,
                 wb + (size_t)(nBase + 32 + l15) * CC,
                 wb + (size_t)(nBase + 48 + l15) * CC,
                 kh, acc);

#pragma unroll
  for (int t = 0; t < 4; ++t) {
    const int col   = nBase + t * 16 + l15;
    const int which = col >> 10;          // 0=Q 1=K 2=V
    const int c     = col & (CC - 1);
    const int h     = c >> 6;
    const int d     = c & (DD - 1);
    const float bv  = bias[col];
#pragma unroll
    for (int mi = 0; mi < 2; ++mi) {
#pragma unroll
      for (int r = 0; r < 8; ++r) {
        const int m  = mBase + mi * 16 + kh * 8 + r;
        const int b  = m >> 11;
        const int t2 = m & (TT - 1);
        const int bh = b * HH + h;
        const unsigned short val = f2bf(acc[mi][t][r] + bv);
        if (which == 0)      Qw[((size_t)bh * TT + t2) * DD + d] = val;
        else if (which == 1) Kw[((size_t)bh * TT + t2) * DD + d] = val;
        else                 Vt[((size_t)bh * DD + d) * TT + t2] = val;
      }
    }
  }
}

// ---------------------------------------------------------------------------
// Flash attention: one wave owns 16 query rows; 8 waves / block (128 queries).
// grid = (T/128, H, B). Causal + padding mask, online softmax, O = P*V.
// Next-chunk K fragments are prefetched under softmax + P*V work.
// ---------------------------------------------------------------------------
__global__ __launch_bounds__(256)
void attn_kernel(const unsigned short* __restrict__ Qw,
                 const unsigned short* __restrict__ Kw,
                 const unsigned short* __restrict__ Vt,
                 const float* __restrict__ amask,
                 unsigned short* __restrict__ AO) {
  __shared__ __align__(16) unsigned short plds[8][16][32];  // per-wave P tile

  const int tid  = threadIdx.x;
  const int w    = tid >> 5;
  const int lane = tid & 31;
  const int l15  = lane & 15;
  const int kh   = lane >> 4;
  const int h = blockIdx.y, b = blockIdx.z;
  const int qBase = blockIdx.x * 128 + w * 16;
  const int bh = b * HH + h;

  const unsigned short* Qb = Qw + (size_t)bh * TT * DD;
  const unsigned short* Kb = Kw + (size_t)bh * TT * DD;
  const unsigned short* Vb = Vt + (size_t)bh * DD * TT;

  // Q fragments for d = [0,32) and [32,64)
  v16bf qf[2];
  const unsigned short* qrow = Qb + (size_t)(qBase + l15) * DD;
#pragma unroll
  for (int kb = 0; kb < 2; ++kb)
    qf[kb] = load_frag16(qrow + kb * 32 + kh * 16, qrow + kb * 32 + kh * 16 + 8);

  v8f oacc[4] = {};
  float runmax[8], runsum[8];
#pragma unroll
  for (int r = 0; r < 8; ++r) { runmax[r] = -1e30f; runsum[r] = 0.0f; }

  auto loadK = [&](v16bf dst[2][2], int kkx) {
#pragma unroll
    for (int nt = 0; nt < 2; ++nt) {
      const unsigned short* krow = Kb + (size_t)(kkx + nt * 16 + l15) * DD + kh * 16;
#pragma unroll
      for (int kb = 0; kb < 2; ++kb)
        dst[nt][kb] = load_frag16(krow + kb * 32, krow + kb * 32 + 8);
    }
  };

  v16bf kf[2][2];
  loadK(kf, 0);

  const int qEnd = qBase + 15;
  for (int kk = 0; kk <= qEnd; kk += 32) {
    // ---- V fragments for this chunk (consumed after softmax) ----
    v16bf vf[4];
#pragma unroll
    for (int dt = 0; dt < 4; ++dt) {
      const unsigned short* vrow = Vb + (size_t)(dt * 16 + l15) * TT + kk + kh * 16;
      vf[dt] = load_frag16(vrow, vrow + 8);
    }
    const float pm0 = amask[b * TT + kk + l15];
    const float pm1 = amask[b * TT + kk + 16 + l15];

    // ---- prefetch next chunk's K fragments ----
    const bool more = (kk + 32) <= qEnd;
    v16bf kfN[2][2];
    if (more) loadK(kfN, kk + 32);

    // ---- scores: two 16x16 key tiles, K-dim = D (2 x 32) ----
    v8f sacc[2] = {};
#pragma unroll
    for (int nt = 0; nt < 2; ++nt)
#pragma unroll
      for (int kb = 0; kb < 2; ++kb)
        sacc[nt] = WMMA_BF16(qf[kb], kf[nt][kb], sacc[nt]);

    float s[2][8];
#pragma unroll
    for (int nt = 0; nt < 2; ++nt) {
      const int   key = kk + nt * 16 + l15;
      const float pm  = nt ? pm1 : pm0;
#pragma unroll
      for (int r = 0; r < 8; ++r) {
        const int q = qBase + kh * 8 + r;
        const float v = sacc[nt][r] * 0.125f;               // 1/sqrt(64)
        s[nt][r] = (key <= q && pm > 0.5f) ? v : -__builtin_inff();
      }
    }

    // ---- online softmax (row = 8*kh + r, spread over 16 lanes) ----
#pragma unroll
    for (int r = 0; r < 8; ++r) {
      float cm = fmaxf(s[0][r], s[1][r]);
#pragma unroll
      for (int off = 1; off < 16; off <<= 1)
        cm = fmaxf(cm, __shfl_xor(cm, off, 32));
      const float nm   = fmaxf(runmax[r], cm);
      const float corr = __expf(runmax[r] - nm);
      runmax[r] = nm;
      const float p0 = __expf(s[0][r] - nm);
      const float p1 = __expf(s[1][r] - nm);
      float ps = p0 + p1;
#pragma unroll
      for (int off = 1; off < 16; off <<= 1)
        ps += __shfl_xor(ps, off, 32);
      runsum[r] = runsum[r] * corr + ps;
#pragma unroll
      for (int t = 0; t < 4; ++t) oacc[t][r] *= corr;
      plds[w][kh * 8 + r][l15]      = f2bf(p0);
      plds[w][kh * 8 + r][16 + l15] = f2bf(p1);
    }

    // same-wave LDS ops are in-order; pin compiler ordering + drain DScnt
    asm volatile("s_wait_dscnt 0x0" ::: "memory");

    // ---- P (A-layout) x V : O(16x64) += P(16x32) * V(32x64) ----
    const unsigned short* prow = &plds[w][l15][0];
    v16bf pf = load_frag16(prow + kh * 16, prow + kh * 16 + 8);
#pragma unroll
    for (int dt = 0; dt < 4; ++dt)
      oacc[dt] = WMMA_BF16(pf, vf[dt], oacc[dt]);

    if (more) {
#pragma unroll
      for (int nt = 0; nt < 2; ++nt)
#pragma unroll
        for (int kb = 0; kb < 2; ++kb)
          kf[nt][kb] = kfN[nt][kb];
    }
  }

  float inv[8];
#pragma unroll
  for (int r = 0; r < 8; ++r) inv[r] = 1.0f / runsum[r];
#pragma unroll
  for (int dt = 0; dt < 4; ++dt) {
    const int c = h * DD + dt * 16 + l15;
#pragma unroll
    for (int r = 0; r < 8; ++r) {
      const int t = qBase + kh * 8 + r;
      AO[((size_t)b * TT + t) * CC + c] = f2bf(oacc[dt][r] * inv[r]);
    }
  }
}

// ---------------------------------------------------------------------------
// Output projection: y[m,n] = sum_c AO[m,c] * Wout[n,c] + b[n]  (fp32 out)
// M=4096, N=1024, K=1024. Wave tile 32M x 64N.
// ---------------------------------------------------------------------------
__global__ __launch_bounds__(256)
void out_gemm(const unsigned short* __restrict__ ao,
              const unsigned short* __restrict__ wb,
              const float* __restrict__ bias,
              float* __restrict__ out) {
  const int tid  = threadIdx.x;
  const int w    = tid >> 5;
  const int lane = tid & 31;
  const int l15  = lane & 15;
  const int kh   = lane >> 4;
  const int nBase = blockIdx.x * 256 + (w >> 1) * 64;
  const int mBase = blockIdx.y * 64  + (w & 1) * 32;

  v8f acc[2][4] = {};
  mma_tile_32x64(ao + (size_t)(mBase + l15) * CC,
                 ao + (size_t)(mBase + 16 + l15) * CC,
                 wb + (size_t)(nBase +  0 + l15) * CC,
                 wb + (size_t)(nBase + 16 + l15) * CC,
                 wb + (size_t)(nBase + 32 + l15) * CC,
                 wb + (size_t)(nBase + 48 + l15) * CC,
                 kh, acc);

#pragma unroll
  for (int t = 0; t < 4; ++t) {
    const int col  = nBase + t * 16 + l15;
    const float bv = bias[col];
#pragma unroll
    for (int mi = 0; mi < 2; ++mi) {
#pragma unroll
      for (int r = 0; r < 8; ++r) {
        const int m = mBase + mi * 16 + kh * 8 + r;
        out[(size_t)m * CC + col] = acc[mi][t][r] + bv;
      }
    }
  }
}

// ---------------------------------------------------------------------------
extern "C" void kernel_launch(void* const* d_in, const int* in_sizes, int n_in,
                              void* d_out, int out_size, void* d_ws, size_t ws_size,
                              hipStream_t stream) {
  const float* x     = (const float*)d_in[0];
  const float* amask = (const float*)d_in[1];
  const float* Wqkv  = (const float*)d_in[2];
  const float* bqkv  = (const float*)d_in[3];
  const float* Wout  = (const float*)d_in[4];
  const float* bout  = (const float*)d_in[5];
  float* out = (float*)d_out;

  char* ws = (char*)d_ws;
  size_t off = 0;
  auto alloc = [&](size_t bytes) -> void* {
    void* p = ws + off;
    off += (bytes + 255) & ~(size_t)255;
    return p;
  };

  unsigned short* xb  = (unsigned short*)alloc((size_t)MTOT * CC * 2);     // 8 MB
  unsigned short* wqb = (unsigned short*)alloc((size_t)N_QKV * CC * 2);    // 6 MB
  unsigned short* wob = (unsigned short*)alloc((size_t)CC * CC * 2);       // 2 MB
  unsigned short* Qw  = (unsigned short*)alloc((size_t)BB * HH * TT * DD * 2);
  unsigned short* Kw  = (unsigned short*)alloc((size_t)BB * HH * TT * DD * 2);
  unsigned short* Vt  = (unsigned short*)alloc((size_t)BB * HH * DD * TT * 2);
  unsigned short* AO  = (unsigned short*)alloc((size_t)MTOT * CC * 2);     // 8 MB
  (void)ws_size; (void)in_sizes; (void)n_in; (void)out_size;

  const int nx = MTOT * CC;        // 4194304
  const int nw = N_QKV * CC;       // 3145728
  const int no = CC * CC;          // 1048576
  cvt_f32_bf16<<<(nx / 4 + 255) / 256, 256, 0, stream>>>(x, xb, nx);
  cvt_f32_bf16<<<(nw / 4 + 255) / 256, 256, 0, stream>>>(Wqkv, wqb, nw);
  cvt_f32_bf16<<<(no / 4 + 255) / 256, 256, 0, stream>>>(Wout, wob, no);

  qkv_gemm<<<dim3(N_QKV / 256, MTOT / 64), 256, 0, stream>>>(xb, wqb, bqkv, Qw, Kw, Vt);

  attn_kernel<<<dim3(TT / 128, HH, BB), 256, 0, stream>>>(Qw, Kw, Vt, amask, AO);

  out_gemm<<<dim3(CC / 256, MTOT / 64), 256, 0, stream>>>(AO, wob, bout, out);
}
